// MemoryModule_18476949307920
// MI455X (gfx1250) — matrix operand order repaired and checked by reading, not compiled, not run on hardware
//
#include <hip/hip_runtime.h>
#include <math.h>

typedef unsigned short u16;
typedef __attribute__((ext_vector_type(16))) __bf16          v16bf;
typedef __attribute__((ext_vector_type(8)))  float           v8f;
typedef __attribute__((ext_vector_type(8)))  unsigned short  u16x8;
typedef __attribute__((ext_vector_type(4)))  unsigned short  u16x4;
typedef __attribute__((ext_vector_type(8)))  float           f32x8;
typedef __attribute__((ext_vector_type(4)))  float           f32x4;
typedef __attribute__((ext_vector_type(4)))  unsigned int    uint32x4;
typedef __attribute__((ext_vector_type(8)))  int             int32x8;
typedef __attribute__((ext_vector_type(4)))  int             int32x4;

#define D_DIM   1024
#define M_DIM   32768
#define T_DIM   4096      /* B*S */
#define H_DIM   8
#define HD_DIM  128
#define KSEL    16
#define NSLICE  32
#define SLICE_COLS 1024
#define NB      4         /* N-tiles register-blocked per wave */

#if __has_builtin(__builtin_amdgcn_tensor_load_to_lds)
#define USE_TDM 1
#else
#define USE_TDM 0
#endif

__device__ inline u16 f2bf(float f) {
  unsigned u = __builtin_bit_cast(unsigned, f);
  u += 0x7fffu + ((u >> 16) & 1u);           // round-to-nearest-even
  return (u16)(u >> 16);
}
__device__ inline float bf2f(u16 h) {
  unsigned u = ((unsigned)h) << 16;
  return __builtin_bit_cast(float, u);
}

union Frag { v16bf bf; u16x8 h[2]; };

// A fragment: 16x32 bf16 tile at (row0,k0) of row-major [*, ld].
// Lane L<16: M=L, K = k0+{0..7, 16..23}; lane L>=16: M=L-16, K = k0+{8..15, 24..31}.
__device__ inline v16bf load_a(const u16* p, int row0, int k0, int ld, int lane) {
  const u16* r = p + (size_t)(row0 + (lane & 15)) * ld + k0 + ((lane & 16) ? 8 : 0);
  Frag f;
  f.h[0] = *(const u16x8*)(r);
  f.h[1] = *(const u16x8*)(r + 16);
  return f.bf;
}

// A fragment from LDS-staged fp32 rows [16][D_DIM], converted bf16 on the fly.
__device__ inline v16bf load_a_lds_f32(const float* lds, int k0, int lane) {
  const float* r = lds + (size_t)(lane & 15) * D_DIM + k0 + ((lane & 16) ? 8 : 0);
  f32x8 lo = *(const f32x8*)(r);
  f32x8 hi = *(const f32x8*)(r + 16);
  Frag f;
#pragma unroll
  for (int i = 0; i < 8; ++i) { f.h[0][i] = f2bf(lo[i]); f.h[1][i] = f2bf(hi[i]); }
  return f.bf;
}

// B fragment: 32x16 bf16 tile at (k0,n0) of a pre-transposed row-major KxN [*, ld].
// Lane L<16: N=n0+L, K = k0+{0..7, 16..23}; lane L>=16: N=n0+L-16, K = k0+{8..15, 24..31}.
__device__ inline v16bf load_b(const u16* p, int k0, int n0, int ld, int lane) {
  const u16* r = p + (size_t)(k0 + ((lane & 16) ? 8 : 0)) * ld + n0 + (lane & 15);
  Frag f;
#pragma unroll
  for (int i = 0; i < 8; ++i) { f.h[0][i] = r[(size_t)i * ld]; f.h[1][i] = r[(size_t)(16 + i) * ld]; }
  return f.bf;
}

// Guarded branchless sorted-insert: static indices only -> register-resident.
__device__ inline void topk_insert(float (&ts)[KSEL], int (&ti)[KSEL], float sc, int idx) {
  if (sc <= ts[KSEL - 1]) return;
#pragma unroll
  for (int i = KSEL - 1; i >= 1; --i) {
    bool g  = sc > ts[i];
    bool gp = sc > ts[i - 1];
    ts[i] = g ? (gp ? ts[i - 1] : sc)  : ts[i];
    ti[i] = g ? (gp ? ti[i - 1] : idx) : ti[i];
  }
  bool g0 = sc > ts[0];
  ti[0] = g0 ? idx : ti[0];
  ts[0] = g0 ? sc  : ts[0];
}

#if USE_TDM
// TDM gather-mode load: 16 rows (16-bit indices) x tile_dim0 fp32 elements -> LDS.
// D# packing per CDNA5 ISA (sect. 8.3-8.7, gather mode).
// 6-arg builtin form (clang-23 / therock-10.0): extra descriptor group zero-filled.
__device__ inline void tdm_gather16_f32(const float* base, const int (&idx)[KSEL],
                                        unsigned lds_byte_addr) {
  unsigned long long ga = (unsigned long long)(size_t)base;
  uint32x4 g0;
  g0[0] = 0x80000001u;                              // gather_mode=1, idx16, count=1
  g0[1] = lds_byte_addr;                            // D#.lds_addr
  g0[2] = (unsigned)(ga & 0xffffffffu);             // global_addr[31:0]
  g0[3] = (unsigned)((ga >> 32) & 0x01ffffffu)      // global_addr[56:32]
        | 0x80000000u;                              // type=2 ("image")
  int32x8 g1;
  g1[0] = 0x00020000;                               // data_size = 4B
  g1[1] = (int)((D_DIM & 0xffffu) << 16);           // tensor_dim0[15:0]
  g1[2] = (int)0x80000000u;                         // tensor_dim1[15:0] = 32768
  g1[3] = (int)((unsigned)D_DIM << 16);             // tile_dim0 = 1024
  g1[4] = KSEL;                                     // tile_dim1 = #valid indices
  g1[5] = D_DIM;                                    // tensor_dim0_stride lo
  g1[6] = 0;
  g1[7] = 0;
  int32x4 g2, g3;
#pragma unroll
  for (int w = 0; w < 4; ++w) {
    g2[w] = (idx[2 * w]     & 0xffff) | (idx[2 * w + 1] << 16);
    g3[w] = (idx[8 + 2 * w] & 0xffff) | (idx[8 + 2 * w + 1] << 16);
  }
  int32x8 g4 = {};                                  // unused extra group
  __builtin_amdgcn_tensor_load_to_lds(g0, g1, g2, g3, g4, 0);
}
#endif

// ---------------- conversion / transpose ----------------
__global__ void cvt_bf16_kernel(const float* __restrict__ in, u16* __restrict__ out, long long n) {
  long long i = (long long)blockIdx.x * blockDim.x + threadIdx.x;
  long long stride = (long long)gridDim.x * blockDim.x;
  for (; i < n; i += stride) out[i] = f2bf(in[i]);
}

// in: [R][C] fp32 row-major -> out: [C][R] bf16 row-major
__global__ void transpose_bf16_kernel(const float* __restrict__ in, u16* __restrict__ out, int R, int C) {
  long long n = (long long)R * C;
  long long i = (long long)blockIdx.x * blockDim.x + threadIdx.x;
  long long stride = (long long)gridDim.x * blockDim.x;
  for (; i < n; i += stride) {
    int r = (int)(i / C), c = (int)(i % C);
    out[(size_t)c * R + r] = f2bf(in[i]);
  }
}

// ---------------- sims + fused partial top-k ----------------
// grid (T/16 row-blocks, NSLICE column slices), 1 wave per block.
// Each wave: 16 rows x 64 cols per iteration (4 accumulators, A reused 4x).
__global__ void __launch_bounds__(32) sims_topk_kernel(
    const u16* __restrict__ qb, const u16* __restrict__ KT,
    float* __restrict__ cand_s, int* __restrict__ cand_i) {
  const int lane = threadIdx.x;
  const int rb = blockIdx.x;
  const int slice = blockIdx.y;
  __shared__ float tile[16][NB * 16 + 4];
  float ts[KSEL]; int ti[KSEL];
#pragma unroll
  for (int i = 0; i < KSEL; ++i) { ts[i] = -3.4e38f; ti[i] = 0; }
  const int r0 = (lane & 16) ? 8 : 0;

  for (int ct4 = 0; ct4 < SLICE_COLS / (NB * 16); ++ct4) {
    const int col = slice * SLICE_COLS + ct4 * (NB * 16);
    if (ct4 + 1 < SLICE_COLS / (NB * 16))   // pull next B block L2 -> WGP$
      __builtin_prefetch(KT + (size_t)lane * M_DIM + col + NB * 16, 0, 3);
    v8f acc[NB] = {};
#pragma unroll 2
    for (int k0 = 0; k0 < D_DIM; k0 += 32) {
      // batch all fragment loads first so the scheduler can use partial waits
      v16bf a = load_a(qb, rb * 16, k0, D_DIM, lane);
      v16bf b0 = load_b(KT, k0, col +  0, M_DIM, lane);
      v16bf b1 = load_b(KT, k0, col + 16, M_DIM, lane);
      v16bf b2 = load_b(KT, k0, col + 32, M_DIM, lane);
      v16bf b3 = load_b(KT, k0, col + 48, M_DIM, lane);
      acc[0] = __builtin_amdgcn_wmma_f32_16x16x32_bf16(false, a, false, b0, (short)0, acc[0], false, false);
      acc[1] = __builtin_amdgcn_wmma_f32_16x16x32_bf16(false, a, false, b1, (short)0, acc[1], false, false);
      acc[2] = __builtin_amdgcn_wmma_f32_16x16x32_bf16(false, a, false, b2, (short)0, acc[2], false, false);
      acc[3] = __builtin_amdgcn_wmma_f32_16x16x32_bf16(false, a, false, b3, (short)0, acc[3], false, false);
    }
#pragma unroll
    for (int nb = 0; nb < NB; ++nb)
#pragma unroll
      for (int v = 0; v < 8; ++v) tile[r0 + v][nb * 16 + (lane & 15)] = acc[nb][v];
    __syncthreads();
    if (lane < 16) {
      for (int n = 0; n < NB * 16; ++n) topk_insert(ts, ti, tile[lane][n], col + n);
    }
    __syncthreads();
  }
  if (lane < 16) {
    size_t base = ((size_t)(rb * 16 + lane) * NSLICE + slice) * KSEL;
#pragma unroll
    for (int i = 0; i < KSEL; ++i) { cand_s[base + i] = ts[i]; cand_i[base + i] = ti[i]; }
  }
}

// ---------------- merge 32 partial top-16 lists per row ----------------
__global__ void topk_merge_kernel(const float* __restrict__ cand_s,
                                  const int* __restrict__ cand_i,
                                  int* __restrict__ topidx) {
  int row = blockIdx.x * blockDim.x + threadIdx.x;
  if (row >= T_DIM) return;
  float ts[KSEL]; int ti[KSEL];
#pragma unroll
  for (int i = 0; i < KSEL; ++i) { ts[i] = -3.4e38f; ti[i] = 0; }
  size_t base = (size_t)row * NSLICE * KSEL;
  for (int j = 0; j < NSLICE * KSEL; ++j) topk_insert(ts, ti, cand_s[base + j], cand_i[base + j]);
#pragma unroll
  for (int i = 0; i < KSEL; ++i) topidx[row * KSEL + i] = ti[i];
}

// ---------------- generic WMMA GEMM: C[T,N] = A(bf16)[T,D] * BT(bf16)[D,N] + bias ----------------
// grid (T/16, N/64): each wave computes 16x64 with 4 accumulators.
__global__ void __launch_bounds__(32) gemm_bias_kernel(
    const u16* __restrict__ A, const u16* __restrict__ BT,
    const float* __restrict__ bias, float* __restrict__ C, int N) {
  const int lane = threadIdx.x;
  const int rb = blockIdx.x, cb4 = blockIdx.y;
  const int col0 = cb4 * (NB * 16);
  v8f acc[NB] = {};
#pragma unroll 2
  for (int k0 = 0; k0 < D_DIM; k0 += 32) {
    v16bf a = load_a(A, rb * 16, k0, D_DIM, lane);
    v16bf b0 = load_b(BT, k0, col0 +  0, N, lane);
    v16bf b1 = load_b(BT, k0, col0 + 16, N, lane);
    v16bf b2 = load_b(BT, k0, col0 + 32, N, lane);
    v16bf b3 = load_b(BT, k0, col0 + 48, N, lane);
    acc[0] = __builtin_amdgcn_wmma_f32_16x16x32_bf16(false, a, false, b0, (short)0, acc[0], false, false);
    acc[1] = __builtin_amdgcn_wmma_f32_16x16x32_bf16(false, a, false, b1, (short)0, acc[1], false, false);
    acc[2] = __builtin_amdgcn_wmma_f32_16x16x32_bf16(false, a, false, b2, (short)0, acc[2], false, false);
    acc[3] = __builtin_amdgcn_wmma_f32_16x16x32_bf16(false, a, false, b3, (short)0, acc[3], false, false);
  }
  const int r0 = (lane & 16) ? 8 : 0;
#pragma unroll
  for (int nb = 0; nb < NB; ++nb) {
    const int col = col0 + nb * 16 + (lane & 15);
    const float bi = bias[col];
#pragma unroll
    for (int v = 0; v < 8; ++v)
      C[(size_t)(rb * 16 + r0 + v) * N + col] = acc[nb][v] + bi;
  }
}

// ---------------- per-token gathered K/V projection + 16-wide attention ----------------
// Dynamic LDS (139840 B): klds[16][1024]f32 | vlds[16][1024]f32 |
//                         kkb[16][128]bf16 | vvb[16][128]bf16 | qs[128]f32 | scs[16]f32
__global__ void __launch_bounds__(32) kv_attn_kernel(
    const float* __restrict__ mem_k, const float* __restrict__ mem_v,
    const u16* __restrict__ WkT, const u16* __restrict__ WvT,
    const float* __restrict__ in_bias, const float* __restrict__ qproj,
    const int* __restrict__ topidx, u16* __restrict__ attn_out) {
  extern __shared__ char smem[];
  float* klds = (float*)smem;                       // 64 KB
  float* vlds = klds + 16 * D_DIM;                  // 64 KB
  u16*   kkb  = (u16*)(vlds + 16 * D_DIM);          //  4 KB
  u16*   vvb  = kkb + 16 * HD_DIM;                  //  4 KB
  float* qs   = (float*)(vvb + 16 * HD_DIM);        // 512 B
  float* scs  = qs + HD_DIM;                        //  64 B

  const int lane = threadIdx.x;
  const int t = blockIdx.x;
  const float* bk = in_bias + D_DIM;
  const float* bv = in_bias + 2 * D_DIM;
  const int r0 = (lane & 16) ? 8 : 0;

  // --- stage the 16 gathered rows (K and V) into LDS, once per token ---
  int idxs[KSEL];
#pragma unroll
  for (int j = 0; j < KSEL; ++j) idxs[j] = topidx[t * KSEL + j];
#if USE_TDM
  // Tensor Data Mover gather: 16 row indices, 1024 fp32 per row, dest LDS.
  tdm_gather16_f32(mem_k, idxs, (unsigned)(size_t)(void*)klds);
  tdm_gather16_f32(mem_v, idxs, (unsigned)(size_t)(void*)vlds);
#if __has_builtin(__builtin_amdgcn_s_wait_tensorcnt)
  __builtin_amdgcn_s_wait_tensorcnt(0);
#else
  asm volatile("s_wait_tensorcnt 0x0" ::: "memory");
#endif
#else
  for (int j = 0; j < KSEL; ++j) {
    const size_t grow = (size_t)idxs[j] * D_DIM;
    for (int d = lane * 4; d < D_DIM; d += 128) {
      *(f32x4*)&klds[j * D_DIM + d] = *(const f32x4*)(mem_k + grow + d);
      *(f32x4*)&vlds[j * D_DIM + d] = *(const f32x4*)(mem_v + grow + d);
    }
  }
#endif
  __syncthreads();

  for (int h = 0; h < H_DIM; ++h) {
    // --- project the 16 staged rows for this head's 128 columns (2 x 64-wide blocks) ---
    for (int ct4 = 0; ct4 < HD_DIM / (NB * 16); ++ct4) {
      const int col = h * HD_DIM + ct4 * (NB * 16);
      v8f ck[NB] = {}, cv[NB] = {};
      for (int k0 = 0; k0 < D_DIM; k0 += 32) {
        v16bf ak = load_a_lds_f32(klds, k0, lane);
#pragma unroll
        for (int nb = 0; nb < NB; ++nb) {
          v16bf b = load_b(WkT, k0, col + nb * 16, D_DIM, lane);
          ck[nb] = __builtin_amdgcn_wmma_f32_16x16x32_bf16(false, ak, false, b, (short)0, ck[nb], false, false);
        }
        v16bf av = load_a_lds_f32(vlds, k0, lane);
#pragma unroll
        for (int nb = 0; nb < NB; ++nb) {
          v16bf b = load_b(WvT, k0, col + nb * 16, D_DIM, lane);
          cv[nb] = __builtin_amdgcn_wmma_f32_16x16x32_bf16(false, av, false, b, (short)0, cv[nb], false, false);
        }
      }
#pragma unroll
      for (int nb = 0; nb < NB; ++nb) {
        const int cc = ct4 * (NB * 16) + nb * 16 + (lane & 15);
        const float bka = bk[h * HD_DIM + cc], bva = bv[h * HD_DIM + cc];
#pragma unroll
        for (int v = 0; v < 8; ++v) {
          kkb[(r0 + v) * HD_DIM + cc] = f2bf(ck[nb][v] + bka);
          vvb[(r0 + v) * HD_DIM + cc] = f2bf(cv[nb][v] + bva);
        }
      }
    }
    for (int d = lane; d < HD_DIM; d += 32) qs[d] = qproj[(size_t)t * D_DIM + h * HD_DIM + d];
    __syncthreads();
    // --- scores: lane j owns kv slot j ---
    if (lane < 16) {
      float s = 0.f;
      for (int d = 0; d < HD_DIM; ++d) s += bf2f(kkb[lane * HD_DIM + d]) * qs[d];
      scs[lane] = s * 0.08838834764831845f;   // 1/sqrt(128)
    }
    __syncthreads();
    // --- softmax over 16 (redundant per lane) + weighted sum of V ---
    float mx = -3.4e38f;
#pragma unroll
    for (int j = 0; j < 16; ++j) mx = fmaxf(mx, scs[j]);
    float w[16]; float den = 0.f;
#pragma unroll
    for (int j = 0; j < 16; ++j) { w[j] = __expf(scs[j] - mx); den += w[j]; }
    const float inv = 1.f / den;
#pragma unroll
    for (int u = 0; u < 4; ++u) {
      const int d = lane * 4 + u;
      float o = 0.f;
#pragma unroll
      for (int j = 0; j < 16; ++j) o += w[j] * bf2f(vvb[j * HD_DIM + d]);
      attn_out[(size_t)t * D_DIM + h * HD_DIM + d] = f2bf(o * inv);
    }
    __syncthreads();
  }
}

extern "C" void kernel_launch(void* const* d_in, const int* in_sizes, int n_in,
                              void* d_out, int out_size, void* d_ws, size_t ws_size,
                              hipStream_t stream) {
  const float* queries = (const float*)d_in[0];
  const float* mem_k   = (const float*)d_in[1];
  const float* mem_v   = (const float*)d_in[2];
  const float* in_w    = (const float*)d_in[3];   // [3D, D]
  const float* in_b    = (const float*)d_in[4];   // [3D]
  const float* out_w   = (const float*)d_in[5];   // [D, D]
  const float* out_b   = (const float*)d_in[6];   // [D]

  char* ws = (char*)d_ws;
  size_t off = 0;
  auto carve = [&](size_t bytes) -> char* {
    char* p = ws + off;
    off += (bytes + 255) & ~(size_t)255;
    return p;
  };
  u16*   qb       = (u16*)  carve((size_t)T_DIM * D_DIM * 2);          //  8 MB
  u16*   KT       = (u16*)  carve((size_t)D_DIM * M_DIM * 2);          // 64 MB (L2-resident)
  u16*   WqT      = (u16*)  carve((size_t)D_DIM * D_DIM * 2);
  u16*   WkT      = (u16*)  carve((size_t)D_DIM * D_DIM * 2);
  u16*   WvT      = (u16*)  carve((size_t)D_DIM * D_DIM * 2);
  u16*   WoT      = (u16*)  carve((size_t)D_DIM * D_DIM * 2);
  float* qproj    = (float*)carve((size_t)T_DIM * D_DIM * 4);          // 16 MB
  float* cand_s   = (float*)carve((size_t)T_DIM * NSLICE * KSEL * 4);  //  8 MB
  int*   cand_i   = (int*)  carve((size_t)T_DIM * NSLICE * KSEL * 4);  //  8 MB
  int*   topidx   = (int*)  carve((size_t)T_DIM * KSEL * 4);
  u16*   attn_out = (u16*)  carve((size_t)T_DIM * D_DIM * 2);          //  8 MB
  if (off > ws_size) return;  // workspace too small: bail deterministically

  // One-time precision/layout conversions.
  cvt_bf16_kernel<<<2048, 256, 0, stream>>>(queries, qb, (long long)T_DIM * D_DIM);
  transpose_bf16_kernel<<<8192, 256, 0, stream>>>(mem_k, KT, M_DIM, D_DIM);
  transpose_bf16_kernel<<<1024, 256, 0, stream>>>(in_w,                     WqT, D_DIM, D_DIM);
  transpose_bf16_kernel<<<1024, 256, 0, stream>>>(in_w + 1 * D_DIM * D_DIM, WkT, D_DIM, D_DIM);
  transpose_bf16_kernel<<<1024, 256, 0, stream>>>(in_w + 2 * D_DIM * D_DIM, WvT, D_DIM, D_DIM);
  transpose_bf16_kernel<<<1024, 256, 0, stream>>>(out_w,                    WoT, D_DIM, D_DIM);

  // Retrieval: 275-GFLOP bf16 WMMA GEMM fused with per-row top-16.
  sims_topk_kernel<<<dim3(T_DIM / 16, NSLICE), 32, 0, stream>>>(qb, KT, cand_s, cand_i);
  topk_merge_kernel<<<(T_DIM + 255) / 256, 256, 0, stream>>>(cand_s, cand_i, topidx);

  // Q projection.
  gemm_bias_kernel<<<dim3(T_DIM / 16, D_DIM / (NB * 16)), 32, 0, stream>>>(qb, WqT, in_b, qproj, D_DIM);

  // Gathered K/V projection (TDM gather -> LDS) + attention.
  kv_attn_kernel<<<T_DIM, 32, 139840, stream>>>(mem_k, mem_v, WkT, WvT, in_b, qproj, topidx, attn_out);

  // Output projection -> d_out (fp32).
  gemm_bias_kernel<<<dim3(T_DIM / 16, D_DIM / (NB * 16)), 32, 0, stream>>>(attn_out, WoT, out_b, (float*)d_out, D_DIM);
}
